// MambaEncoder_71906342470085
// MI455X (gfx1250) — matrix operand order repaired and checked by reading, not compiled, not run on hardware
//
#include <hip/hip_runtime.h>
#include <hip/hip_bf16.h>

// ---------------- model dims ----------------
#define BSZ    4
#define LSEQ   512
#define NT     (BSZ * LSEQ)      // 2048 tokens
#define DM     768
#define DI     1536
#define NSTATE 16
#define DTRANK 48
#define DTPAD  64                // dt_rank zero-padded to K%32==0
#define DBC_W  80                // dt_rank + 2*N
#define FS_DT  (1.0f / 30.0f)

typedef __attribute__((ext_vector_type(16))) __bf16 v16bf;
typedef __attribute__((ext_vector_type(8)))  float  v8f;

__device__ __forceinline__ float siluf(float v)     { return v / (1.f + __expf(-v)); }
__device__ __forceinline__ float softplusf(float v) { return (v > 20.f) ? v : __logf(1.f + __expf(v)); }

// =====================================================================
// WMMA GEMM:  C[M,N] = act(A[M,K] @ W[N,K]^T + bias) (+ residual)
// bf16 in-register convert, f32 accumulate via v_wmma_f32_16x16x32_bf16.
// block = 256 threads = 8 waves (4M x 2N); block tile 128x128;
// wave tile 32(M) x 64(N) = 2 A-frags + 4 B-frags -> 8 WMMA / K-step.
// Templated on <ACT, HAS_BIAS, HAS_RES> so the epilogue is branch-free.
// REQUIREMENTS (hold for every call site):
//   * K % 32 == 0 (dt_proj weights are zero-padded to 64)
//   * out-of-range M/N load rows are CLAMPED (garbage only reaches
//     output rows/cols that the guarded store never writes)
// =====================================================================
__device__ __forceinline__ void load8u(const float* p, int k0, v16bf& f, int base) {
  const float4* q = (const float4*)(p + k0);
  float4 u0 = q[0], u1 = q[1];
  f[base + 0] = (__bf16)u0.x; f[base + 1] = (__bf16)u0.y;
  f[base + 2] = (__bf16)u0.z; f[base + 3] = (__bf16)u0.w;
  f[base + 4] = (__bf16)u1.x; f[base + 5] = (__bf16)u1.y;
  f[base + 6] = (__bf16)u1.z; f[base + 7] = (__bf16)u1.w;
}

// A fragment 16x32 (MxK): lanes 0-15 row m, K {0..7,16..23}; lanes 16-31 K {8..15,24..31}
__device__ __forceinline__ v16bf fragA(const float* A, int lda, int m0, int M,
                                       int kk, int lane) {
  v16bf f;
  int r  = m0 + (lane & 15); r = (r < M) ? r : (M - 1);   // clamp, no branch
  int hi = (lane >> 4) & 1;
  const float* p = A + (size_t)r * lda;
  load8u(p, kk + hi * 8,      f, 0);
  load8u(p, kk + 16 + hi * 8, f, 8);
  return f;
}

// B fragment 32x16 (KxN): lane n holds col n; lanes 0-15 K {0..7,16..23}; lanes 16-31 K {8..15,24..31}
// column n of B == row n of W (W is [N,K] row-major)
__device__ __forceinline__ v16bf fragB(const float* W, int ldb, int n0, int N,
                                       int kk, int lane) {
  v16bf f;
  int n  = n0 + (lane & 15); n = (n < N) ? n : (N - 1);   // clamp, no branch
  int hi = (lane >> 4) & 1;
  const float* p = W + (size_t)n * ldb;
  load8u(p, kk + hi * 8,      f, 0);
  load8u(p, kk + 16 + hi * 8, f, 8);
  return f;
}

// C/D layout: lane 0-15 -> N=lane (M=r+0..7), lane 16-31 -> (M=r+8..15)
template<int ACT, bool HAS_BIAS, bool HAS_RES>
__device__ __forceinline__ void store_tile(v8f acc, int m0, int n0, int lane,
                                           const float* bias,
                                           const float* resid, int ldr,
                                           float* C, int ldc,
                                           int M, int N) {
  int col = n0 + (lane & 15);
  if (col >= N) return;
  int hi = (lane >> 4) & 1;
  float bv = HAS_BIAS ? bias[col] : 0.f;
#pragma unroll
  for (int r = 0; r < 8; ++r) {
    int row = m0 + r + hi * 8;
    if (row >= M) continue;
    float v = acc[r] + bv;
    if (ACT == 1)      v = siluf(v);
    else if (ACT == 2) v = softplusf(v);
    if (HAS_RES) v += resid[(size_t)row * ldr + col];
    C[(size_t)row * ldc + col] = v;
  }
}

template<int ACT, bool HAS_BIAS, bool HAS_RES>
__global__ __launch_bounds__(256) void k_gemm(
    const float* __restrict__ A, int lda,
    const float* __restrict__ W, int ldb,
    const float* __restrict__ bias,
    const float* resid, int ldr,
    float* C, int ldc,
    int M, int N, int K) {
  int lane = threadIdx.x & 31;
  int wave = threadIdx.x >> 5;
  int wm = wave >> 1, wn = wave & 1;
  int m0 = blockIdx.y * 128 + wm * 32;
  int n0 = blockIdx.x * 128 + wn * 64;

  v8f z8 = {0.f,0.f,0.f,0.f,0.f,0.f,0.f,0.f};
  v8f acc[2][4] = {{z8,z8,z8,z8},{z8,z8,z8,z8}};

  for (int kk = 0; kk < K; kk += 32) {
    v16bf fa0 = fragA(A, lda, m0,      M, kk, lane);
    v16bf fa1 = fragA(A, lda, m0 + 16, M, kk, lane);
    v16bf fb0 = fragB(W, ldb, n0,      N, kk, lane);
    v16bf fb1 = fragB(W, ldb, n0 + 16, N, kk, lane);
    v16bf fb2 = fragB(W, ldb, n0 + 32, N, kk, lane);
    v16bf fb3 = fragB(W, ldb, n0 + 48, N, kk, lane);
    acc[0][0] = __builtin_amdgcn_wmma_f32_16x16x32_bf16(false, fa0, false, fb0, (short)0, acc[0][0], false, false);
    acc[0][1] = __builtin_amdgcn_wmma_f32_16x16x32_bf16(false, fa0, false, fb1, (short)0, acc[0][1], false, false);
    acc[0][2] = __builtin_amdgcn_wmma_f32_16x16x32_bf16(false, fa0, false, fb2, (short)0, acc[0][2], false, false);
    acc[0][3] = __builtin_amdgcn_wmma_f32_16x16x32_bf16(false, fa0, false, fb3, (short)0, acc[0][3], false, false);
    acc[1][0] = __builtin_amdgcn_wmma_f32_16x16x32_bf16(false, fa1, false, fb0, (short)0, acc[1][0], false, false);
    acc[1][1] = __builtin_amdgcn_wmma_f32_16x16x32_bf16(false, fa1, false, fb1, (short)0, acc[1][1], false, false);
    acc[1][2] = __builtin_amdgcn_wmma_f32_16x16x32_bf16(false, fa1, false, fb2, (short)0, acc[1][2], false, false);
    acc[1][3] = __builtin_amdgcn_wmma_f32_16x16x32_bf16(false, fa1, false, fb3, (short)0, acc[1][3], false, false);
  }
#pragma unroll
  for (int i = 0; i < 2; ++i)
#pragma unroll
    for (int j = 0; j < 4; ++j)
      store_tile<ACT, HAS_BIAS, HAS_RES>(acc[i][j], m0 + i * 16, n0 + j * 16, lane,
                                         bias, resid, ldr, C, ldc, M, N);
}

// =====================================================================
// zero-pad dt_proj_w [DI,48] -> [DI,64]
// =====================================================================
__global__ __launch_bounds__(256) void k_pad_dtw(const float* __restrict__ w,
                                                 float* __restrict__ wp) {
  int i = blockIdx.x * blockDim.x + threadIdx.x;
  if (i >= DI * DTPAD) return;
  int row = i / DTPAD, col = i % DTPAD;
  wp[i] = (col < DTRANK) ? w[row * DTRANK + col] : 0.f;
}

// =====================================================================
// LayerNorm over last dim (768); one block per token
// =====================================================================
__global__ __launch_bounds__(256) void k_layernorm(const float* __restrict__ x,
                                                   const float* __restrict__ w,
                                                   const float* __restrict__ b,
                                                   float* __restrict__ out) {
  __shared__ float sm[8];
  __shared__ float stat[2];
  int row = blockIdx.x;
  const float* p = x + (size_t)row * DM;
  int tid = threadIdx.x, lane = tid & 31, wid = tid >> 5;

  float s = 0.f;
  for (int i = tid; i < DM; i += 256) s += p[i];
#pragma unroll
  for (int o = 16; o > 0; o >>= 1) s += __shfl_down(s, o, 32);
  if (lane == 0) sm[wid] = s;
  __syncthreads();
  if (tid == 0) {
    float t = 0.f;
    for (int i = 0; i < 8; ++i) t += sm[i];
    stat[0] = t / (float)DM;
  }
  __syncthreads();
  float mean = stat[0];

  float v = 0.f;
  for (int i = tid; i < DM; i += 256) { float d = p[i] - mean; v += d * d; }
#pragma unroll
  for (int o = 16; o > 0; o >>= 1) v += __shfl_down(v, o, 32);
  if (lane == 0) sm[wid] = v;
  __syncthreads();
  if (tid == 0) {
    float t = 0.f;
    for (int i = 0; i < 8; ++i) t += sm[i];
    stat[1] = rsqrtf(t / (float)DM + 1e-5f);
  }
  __syncthreads();
  float inv = stat[1];
  float* o = out + (size_t)row * DM;
  for (int i = tid; i < DM; i += 256)
    o[i] = (p[i] - mean) * inv * w[i] + b[i];
}

// =====================================================================
// causal depthwise conv (k=4) + SiLU ; xz row stride 3072, xc row stride 1536
// =====================================================================
__global__ __launch_bounds__(256) void k_conv_silu(const float* __restrict__ xz,
                                                   const float* __restrict__ cw,
                                                   const float* __restrict__ cb,
                                                   float* __restrict__ xc) {
  int row = blockIdx.x;              // b*512 + t
  int b = row >> 9, t = row & 511;
  for (int c = threadIdx.x; c < DI; c += 256) {
    float acc = cb[c];
#pragma unroll
    for (int j = 0; j < 4; ++j) {
      int tt = t - 3 + j;
      if (tt >= 0)
        acc += cw[c * 4 + j] * xz[((size_t)(b * LSEQ + tt)) * (2 * DI) + c];
    }
    xc[(size_t)row * DI + c] = siluf(acc);
  }
}

// =====================================================================
// selective scan: one lane per (b, d_inner); 16 states in VGPRs
// =====================================================================
__global__ __launch_bounds__(256) void k_scan(const float* __restrict__ delta,
                                              const float* __restrict__ xc,
                                              const float* __restrict__ dbc,
                                              const float* __restrict__ A_log,
                                              const float* __restrict__ Dp,
                                              float* __restrict__ y) {
  int idx = blockIdx.x * blockDim.x + threadIdx.x;  // b*DI + d
  if (idx >= BSZ * DI) return;
  int b = idx / DI, d = idx % DI;
  float a[NSTATE], h[NSTATE];
#pragma unroll
  for (int n = 0; n < NSTATE; ++n) { a[n] = -__expf(A_log[d * NSTATE + n]); h[n] = 0.f; }
  float Dd = Dp[d];
  for (int t = 0; t < LSEQ; ++t) {
    size_t rb = (size_t)(b * LSEQ + t);
    float dl = delta[rb * DI + d];
    float u  = xc[rb * DI + d];
    const float* bc = dbc + rb * DBC_W;
    float acc = 0.f;
#pragma unroll
    for (int n = 0; n < NSTATE; ++n) {
      float Bn = bc[DTRANK + n];
      float Cn = bc[DTRANK + NSTATE + n];
      h[n] = __expf(dl * a[n]) * h[n] + dl * Bn * u;
      acc += h[n] * Cn;
    }
    y[rb * DI + d] = acc + u * Dd;
  }
}

// =====================================================================
// y *= silu(gate), gate = xz[:, DI:2*DI]
// =====================================================================
__global__ __launch_bounds__(256) void k_gatemul(float* __restrict__ y,
                                                 const float* __restrict__ xz) {
  int i = blockIdx.x * blockDim.x + threadIdx.x;
  if (i >= NT * DI) return;
  int row = i / DI, col = i % DI;
  y[i] *= siluf(xz[(size_t)row * (2 * DI) + DI + col]);
}

// =====================================================================
// LTI band filter: one lane per (b, d_model); 16 states in VGPRs
// =====================================================================
__global__ __launch_bounds__(256) void k_band(const float* __restrict__ z,
                                              const float* __restrict__ A_log,
                                              const float* __restrict__ Bp,
                                              const float* __restrict__ Cp,
                                              float* __restrict__ out) {
  int idx = blockIdx.x * blockDim.x + threadIdx.x;  // b*DM + d
  if (idx >= BSZ * DM) return;
  int b = idx / DM, d = idx % DM;
  float dA[NSTATE], dB[NSTATE], Cc[NSTATE], h[NSTATE];
#pragma unroll
  for (int n = 0; n < NSTATE; ++n) {
    dA[n] = __expf(-__expf(A_log[d * NSTATE + n]) * FS_DT);
    dB[n] = FS_DT * Bp[d * NSTATE + n];
    Cc[n] = Cp[d * NSTATE + n];
    h[n] = 0.f;
  }
  for (int t = 0; t < LSEQ; ++t) {
    size_t rb = (size_t)(b * LSEQ + t);
    float x = z[rb * DM + d];
    float acc = 0.f;
#pragma unroll
    for (int n = 0; n < NSTATE; ++n) {
      h[n] = dA[n] * h[n] + dB[n] * x;
      acc += h[n] * Cc[n];
    }
    out[rb * DM + d] = acc;
  }
}

// =====================================================================
// concat(ssm_out, 0.5*(hr+br)) -> cbuf ;  hr_sum += hr ; br_sum += br
// =====================================================================
__global__ __launch_bounds__(256) void k_combine(const float* __restrict__ ssm,
                                                 const float* __restrict__ hr,
                                                 const float* __restrict__ br,
                                                 float* __restrict__ cbuf,
                                                 float* __restrict__ hr_sum,
                                                 float* __restrict__ br_sum) {
  int i = blockIdx.x * blockDim.x + threadIdx.x;
  if (i >= NT * DM) return;
  int row = i / DM, c = i % DM;
  float hv = hr[i], bv = br[i];
  cbuf[(size_t)row * (2 * DM) + c]      = ssm[i];
  cbuf[(size_t)row * (2 * DM) + DM + c] = 0.5f * (hv + bv);
  hr_sum[i] += hv;
  br_sum[i] += bv;
}

__global__ __launch_bounds__(256) void k_copy(float* __restrict__ dst,
                                              const float* __restrict__ src, int n) {
  int i = blockIdx.x * blockDim.x + threadIdx.x;
  if (i < n) dst[i] = src[i];
}
__global__ __launch_bounds__(256) void k_zero(float* __restrict__ dst, int n) {
  int i = blockIdx.x * blockDim.x + threadIdx.x;
  if (i < n) dst[i] = 0.f;
}
__global__ __launch_bounds__(256) void k_scale(float* __restrict__ dst,
                                               const float* __restrict__ src,
                                               float s, int n) {
  int i = blockIdx.x * blockDim.x + threadIdx.x;
  if (i < n) dst[i] = src[i] * s;
}

// =====================================================================
// host orchestration
// =====================================================================
// act: 0 = none, 1 = silu, 2 = softplus
static void gemm(hipStream_t s, const float* A, int lda, const float* W, int ldb,
                 const float* bias, const float* res, int ldr,
                 float* C, int ldc, int M, int N, int K, int act) {
  dim3 g((N + 127) / 128, (M + 127) / 128);
  if (res != nullptr) {            // bias + residual, no act
    k_gemm<0, true, true><<<g, 256, 0, s>>>(A, lda, W, ldb, bias, res, ldr, C, ldc, M, N, K);
  } else if (bias == nullptr) {    // plain
    k_gemm<0, false, false><<<g, 256, 0, s>>>(A, lda, W, ldb, nullptr, nullptr, 0, C, ldc, M, N, K);
  } else if (act == 1) {           // bias + silu
    k_gemm<1, true, false><<<g, 256, 0, s>>>(A, lda, W, ldb, bias, nullptr, 0, C, ldc, M, N, K);
  } else if (act == 2) {           // bias + softplus
    k_gemm<2, true, false><<<g, 256, 0, s>>>(A, lda, W, ldb, bias, nullptr, 0, C, ldc, M, N, K);
  } else {                         // bias only
    k_gemm<0, true, false><<<g, 256, 0, s>>>(A, lda, W, ldb, bias, nullptr, 0, C, ldc, M, N, K);
  }
}

struct LayerPtrs {
  const float *nw, *nb;
  const float *ipw, *ipb, *cw, *cb, *xpw, *dpw, *dpb, *Alog, *Dp, *opw, *opb;
  const float *hrA, *hrB, *hrC, *brA, *brB, *brC, *f1w, *f1b, *f2w, *f2b;
  bool freq;
};

extern "C" void kernel_launch(void* const* d_in, const int* in_sizes, int n_in,
                              void* d_out, int out_size, void* d_ws, size_t ws_size,
                              hipStream_t stream) {
  (void)in_sizes; (void)n_in; (void)out_size; (void)ws_size;
  int idx = 0;
  auto nf = [&]() -> const float* { return (const float*)d_in[idx++]; };

  const float* x = nf();
  LayerPtrs L[4];
  for (int l = 0; l < 4; ++l) {
    L[l].nw = nf(); L[l].nb = nf();
    L[l].ipw = nf(); L[l].ipb = nf(); L[l].cw = nf(); L[l].cb = nf();
    L[l].xpw = nf(); L[l].dpw = nf(); L[l].dpb = nf();
    L[l].Alog = nf(); L[l].Dp = nf(); L[l].opw = nf(); L[l].opb = nf();
    L[l].freq = (l > 0);
    if (l > 0) {
      L[l].hrA = nf(); L[l].hrB = nf(); L[l].hrC = nf();
      L[l].brA = nf(); L[l].brB = nf(); L[l].brC = nf();
      L[l].f1w = nf(); L[l].f1b = nf(); L[l].f2w = nf(); L[l].f2b = nf();
    }
  }
  const float* fnw = nf();
  const float* fnb = nf();

  // ---- workspace carve-up (floats) ----
  float* ws = (float*)d_ws;
  float* cur     = ws; ws += NT * DM;
  float* z       = ws; ws += NT * DM;
  float* xz      = ws; ws += NT * 2 * DI;
  float* xc      = ws; ws += NT * DI;
  float* dbc     = ws; ws += NT * DBC_W;
  float* delta   = ws; ws += NT * DI;
  float* ybuf    = ws; ws += NT * DI;
  float* ssm_out = ws; ws += NT * DM;
  float* cbuf    = ws; ws += NT * 2 * DM;
  float* t1      = ws; ws += NT * DM;
  float* hr_sum  = ws; ws += NT * DM;
  float* br_sum  = ws; ws += NT * DM;
  float* hr_tmp  = ws; ws += NT * DM;
  float* br_tmp  = ws; ws += NT * DM;
  float* dpw_pad = ws; ws += DI * DTPAD;

  const int nElemDM = NT * DM;       // 1,572,864
  const int nElemDI = NT * DI;       // 3,145,728
  dim3 bDM((nElemDM + 255) / 256), bDI((nElemDI + 255) / 256);

  k_copy<<<bDM, 256, 0, stream>>>(cur, x, nElemDM);
  k_zero<<<bDM, 256, 0, stream>>>(hr_sum, nElemDM);
  k_zero<<<bDM, 256, 0, stream>>>(br_sum, nElemDM);

  for (int l = 0; l < 4; ++l) {
    const LayerPtrs& p = L[l];
    // 0. zero-padded dt_proj weights (K 48 -> 64 so GEMM K%32==0)
    k_pad_dtw<<<(DI * DTPAD + 255) / 256, 256, 0, stream>>>(p.dpw, dpw_pad);
    // 1. z = LN(cur)
    k_layernorm<<<NT, 256, 0, stream>>>(cur, p.nw, p.nb, z);
    // 2. xz = z @ in_proj_w^T + b           [2048, 3072]
    gemm(stream, z, DM, p.ipw, DM, p.ipb, nullptr, 0, xz, 2 * DI, NT, 2 * DI, DM, 0);
    // 3. causal depthwise conv + silu -> xc [2048, 1536]
    k_conv_silu<<<NT, 256, 0, stream>>>(xz, p.cw, p.cb, xc);
    // 4. dbc = xc @ x_proj_w^T              [2048, 80]
    gemm(stream, xc, DI, p.xpw, DI, nullptr, nullptr, 0, dbc, DBC_W, NT, DBC_W, DI, 0);
    // 5. delta = softplus(dbc[:, :64] @ dpw_pad^T + b)  [2048, 1536]
    //    (cols 48..63 of dbc hit zero weight columns -> no contribution)
    gemm(stream, dbc, DBC_W, dpw_pad, DTPAD, p.dpb, nullptr, 0, delta, DI, NT, DI, DTPAD, 2);
    // 6. selective scan -> ybuf
    k_scan<<<(BSZ * DI + 255) / 256, 256, 0, stream>>>(delta, xc, dbc, p.Alog, p.Dp, ybuf);
    // 7. ybuf *= silu(gate)
    k_gatemul<<<bDI, 256, 0, stream>>>(ybuf, xz);
    // 8. out_proj (+ residual straight into cur for non-freq layer)
    if (!p.freq) {
      gemm(stream, ybuf, DI, p.opw, DI, p.opb, cur, DM, cur, DM, NT, DM, DI, 0);
    } else {
      gemm(stream, ybuf, DI, p.opw, DI, p.opb, nullptr, 0, ssm_out, DM, NT, DM, DI, 0);
      // 9a. band filters on z
      k_band<<<(BSZ * DM + 255) / 256, 256, 0, stream>>>(z, p.hrA, p.hrB, p.hrC, hr_tmp);
      k_band<<<(BSZ * DM + 255) / 256, 256, 0, stream>>>(z, p.brA, p.brB, p.brC, br_tmp);
      // 9b. concat + accumulate band sums
      k_combine<<<bDM, 256, 0, stream>>>(ssm_out, hr_tmp, br_tmp, cbuf, hr_sum, br_sum);
      // 9c. t1 = silu(cbuf @ fus1_w^T + b1)
      gemm(stream, cbuf, 2 * DM, p.f1w, 2 * DM, p.f1b, nullptr, 0, t1, DM, NT, DM, 2 * DM, 1);
      // 9d. cur = t1 @ fus2_w^T + b2 + cur
      gemm(stream, t1, DM, p.f2w, DM, p.f2b, cur, DM, cur, DM, NT, DM, DM, 0);
    }
  }

  // final norm + band means
  float* out = (float*)d_out;
  k_layernorm<<<NT, 256, 0, stream>>>(cur, fnw, fnb, out);
  k_scale<<<bDM, 256, 0, stream>>>(out + (size_t)NT * DM,     hr_sum, 1.f / 3.f, nElemDM);
  k_scale<<<bDM, 256, 0, stream>>>(out + (size_t)2 * NT * DM, br_sum, 1.f / 3.f, nElemDM);
}
